// Sequence_1803886265571
// MI455X (gfx1250) — matrix operand order repaired and checked by reading, not compiled, not run on hardware
//
#include <hip/hip_runtime.h>

typedef __attribute__((ext_vector_type(16))) _Float16 v16h;
typedef __attribute__((ext_vector_type(8)))  _Float16 v8h;
typedef __attribute__((ext_vector_type(8)))  float    v8f;

namespace {
constexpr int kH1   = 51;        // lstm1 hidden
constexpr int kNG   = 4 * kH1;   // 204 gate columns
constexpr int kNT   = 13;        // ceil(204/16) WMMA N tiles (pad to 208)
constexpr int kKC   = 2;         // K chunks of 32 (51 padded to 64)
constexpr int kRows = 16;        // batch rows per workgroup = one WMMA M tile
constexpr int kGS   = 210;       // gates LDS row stride (floats), bank-conflict-free C stores
constexpr int kH1K  = 64;        // padded K extent of h1 staging
constexpr int kFrag = 512;       // halfs per B fragment (32 lanes x 16)
constexpr int kB    = 2048;      // batch (fixed by reference setup)
}

__device__ __forceinline__ float sigm(float x)      { return 1.0f / (1.0f + __expf(-x)); }
__device__ __forceinline__ float tanh_fast(float x) { return 1.0f - 2.0f / (1.0f + __expf(2.0f * x)); }

__global__ __launch_bounds__(32) void lstm_seq_kernel(
    const float* __restrict__ input,   // [B][T]
    const float* __restrict__ Wih1,    // [204][1]
    const float* __restrict__ Whh1,    // [204][51]
    const float* __restrict__ bih1,    // [204]
    const float* __restrict__ bhh1,    // [204]
    const float* __restrict__ Wih2,    // [4][51]
    const float* __restrict__ Whh2,    // [4][1]
    const float* __restrict__ bih2,    // [4]
    const float* __restrict__ bhh2,    // [4]
    const int*   __restrict__ futurep, // scalar
    float*       __restrict__ out,     // [B][T+F]
    int T)
{
  __shared__ __attribute__((aligned(32))) _Float16 ldsB[kKC * kNT * kFrag]; // W_hh1^T fragments
  __shared__ __attribute__((aligned(16))) _Float16 h1buf[kRows * kH1K];     // h1 state (f16, K-padded)
  __shared__ float gates[kRows * kGS];   // WMMA C staging
  __shared__ float biassum[kNG];         // b_ih1 + b_hh1
  __shared__ float wih1c[kNG];           // W_ih1[:,0]
  __shared__ float wih2s[4][kH1 + 1];    // W_ih2 rows (padded)
  __shared__ float bias2s[4];
  __shared__ float whh2s[4];
  __shared__ float xbuf[kRows];          // per-row x_t (input, or fed-back h2)

  const int tid = threadIdx.x;           // 0..31, one wave
  const int b0  = blockIdx.x * kRows;
  const int F   = *futurep;
  const int TT  = T + F;

  // ---------------- one-time staging ----------------
  for (int j = tid; j < kNG; j += 32) {
    biassum[j] = bih1[j] + bhh1[j];
    wih1c[j]   = Wih1[j];
  }
  for (int j = tid; j < 4 * kH1; j += 32) wih2s[j / kH1][j % kH1] = Wih2[j];
  if (tid < 4) { bias2s[tid] = bih2[tid] + bhh2[tid]; whh2s[tid] = Whh2[tid]; }
  for (int j = tid; j < kRows * kH1K; j += 32) h1buf[j] = (_Float16)0.0f; // incl. K padding

  const int hi16  = (tid >= 16);
  const int nlane = tid & 15;

  // Stage W_hh1^T (K x N, f16, zero-padded) in exact B-fragment per-lane register order:
  // lane l (l<16): N = nt*16 + l, halfs h=0..7 -> K = kc*32 + h, h=8..15 -> K = kc*32 + 16 + (h-8) ; lanes 16-31: same with K += 8.
  for (int f = 0; f < kKC * kNT; ++f) {
    const int kc = f / kNT, nt = f % kNT;
    const int n  = nt * 16 + nlane;
    for (int h = 0; h < 16; ++h) {
      const int k = kc * 32 + ((h & 8) ? 16 : 0) + (hi16 ? 8 : 0) + (h & 7);
      const float v = (n < kNG && k < kH1) ? Whh1[n * kH1 + k] : 0.0f;
      ldsB[f * kFrag + tid * 16 + h] = (_Float16)v;
    }
  }

  // Per-thread recurrent state.
  float c1v[26];
  #pragma unroll
  for (int k = 0; k < 26; ++k) c1v[k] = 0.0f;
  float h2 = 0.0f, c2 = 0.0f;

  const int em   = tid & 15;            // elementwise: batch row owned by this lane
  const int en0  = tid >> 4;            // elementwise: hidden-column parity
  const int am   = tid & 15;            // A fragment: M = lane%16
  const int koff = hi16 ? 8 : 0;        // A fragment K sub-offset per ISA layout
  const int mofs = hi16 ? 8 : 0;        // C fragment M offset per ISA layout

  __syncthreads();

  for (int t = 0; t < TT; ++t) {
    // Stage x_t: main phase reads input; future phase uses h2 left in xbuf by previous step.
    if (t < T && tid < kRows) xbuf[tid] = input[(size_t)(b0 + tid) * T + t];

    // ---------------- phase 1: gates_pre = h1 @ W_hh1^T via WMMA ----------------
    v8f acc[kNT];
    #pragma unroll
    for (int nt = 0; nt < kNT; ++nt) {
      #pragma unroll
      for (int r = 0; r < 8; ++r) acc[nt][r] = 0.0f;
    }
    #pragma unroll
    for (int kc = 0; kc < kKC; ++kc) {
      const int kb = kc * 32;
      // A fragment (16x32 f16): two aligned 8-half runs per lane.
      const v8h a0 = *(const v8h*)&h1buf[am * kH1K + kb + koff];
      const v8h a1 = *(const v8h*)&h1buf[am * kH1K + kb + 16 + koff];
      v16h A;
      #pragma unroll
      for (int i = 0; i < 8; ++i) { A[i] = a0[i]; A[i + 8] = a1[i]; }
      #pragma unroll
      for (int nt = 0; nt < kNT; ++nt) {
        const v16h Bf = *(const v16h*)&ldsB[(kc * kNT + nt) * kFrag + tid * 16];
        acc[nt] = __builtin_amdgcn_wmma_f32_16x16x32_f16(
            /*neg_a=*/false, A, /*neg_b=*/false, Bf,
            /*c_mod=*/(short)0, acc[nt], /*reuse_a=*/false, /*reuse_b=*/false);
      }
    }
    // Scatter C tiles to LDS (lane<16: M=r,N=lane ; lane>=16: M=r+8,N=lane-16).
    #pragma unroll
    for (int nt = 0; nt < kNT; ++nt) {
      #pragma unroll
      for (int r = 0; r < 8; ++r)
        gates[(r + mofs) * kGS + nt * 16 + nlane] = acc[nt][r];
    }
    __syncthreads();

    // ---------------- phase 2: gate nonlinearities, c1/h1 update ----------------
    const float xm = xbuf[em];
    #pragma unroll
    for (int k = 0; k < 26; ++k) {
      const int n = en0 + 2 * k;
      if (n < kH1) {
        const int ji = n, jf = n + kH1, jg = n + 2 * kH1, jo = n + 3 * kH1;
        const float gi = gates[em * kGS + ji] + biassum[ji] + xm * wih1c[ji];
        const float gf = gates[em * kGS + jf] + biassum[jf] + xm * wih1c[jf];
        const float gg = gates[em * kGS + jg] + biassum[jg] + xm * wih1c[jg];
        const float go = gates[em * kGS + jo] + biassum[jo] + xm * wih1c[jo];
        const float c  = sigm(gf) * c1v[k] + sigm(gi) * tanh_fast(gg);
        c1v[k] = c;
        h1buf[em * kH1K + n] = (_Float16)(sigm(go) * tanh_fast(c));
      }
    }
    __syncthreads();

    // ---------------- phase 3: LSTM2 (H2=1), output, feed-back ----------------
    if (tid < kRows) {
      float d0 = bias2s[0] + h2 * whh2s[0];
      float d1 = bias2s[1] + h2 * whh2s[1];
      float d2 = bias2s[2] + h2 * whh2s[2];
      float d3 = bias2s[3] + h2 * whh2s[3];
      #pragma unroll
      for (int k = 0; k < kH1; ++k) {
        const float hv = (float)h1buf[tid * kH1K + k];
        d0 += hv * wih2s[0][k]; d1 += hv * wih2s[1][k];
        d2 += hv * wih2s[2][k]; d3 += hv * wih2s[3][k];
      }
      c2 = sigm(d1) * c2 + sigm(d0) * tanh_fast(d2);
      h2 = sigm(d3) * tanh_fast(c2);
      out[(size_t)(b0 + tid) * TT + t] = h2;
      xbuf[tid] = h2;   // x_{t+1} during the future phase
    }
    __syncthreads();
  }
}

extern "C" void kernel_launch(void* const* d_in, const int* in_sizes, int n_in,
                              void* d_out, int out_size, void* d_ws, size_t ws_size,
                              hipStream_t stream) {
  (void)n_in; (void)out_size; (void)d_ws; (void)ws_size;
  const float* input = (const float*)d_in[0];
  const float* Wih1  = (const float*)d_in[1];
  const float* Whh1  = (const float*)d_in[2];
  const float* bih1  = (const float*)d_in[3];
  const float* bhh1  = (const float*)d_in[4];
  const float* Wih2  = (const float*)d_in[5];
  const float* Whh2  = (const float*)d_in[6];
  const float* bih2  = (const float*)d_in[7];
  const float* bhh2  = (const float*)d_in[8];
  const int*   fut   = (const int*)d_in[9];
  const int T = in_sizes[0] / kB;   // B = 2048 per reference setup

  dim3 grid(kB / kRows), block(32);
  lstm_seq_kernel<<<grid, block, 0, stream>>>(
      input, Wih1, Whh1, bih1, bhh1, Wih2, Whh2, bih2, bhh2, fut,
      (float*)d_out, T);
}
